// QueryNet_38568806318522
// MI455X (gfx1250) — matrix operand order repaired and checked by reading, not compiled, not run on hardware
//
#include <hip/hip_runtime.h>
#include <hip/hip_bf16.h>
#include <math.h>

// ---------------------------------------------------------------------------
// Problem constants (from reference)
// ---------------------------------------------------------------------------
#define PAD_ID   40001
#define DIM_EMB  300
#define DIM_EMB_PAD 320          // padded to multiple of 32 for WMMA K
#define DIM_H    512
#define BATCH    256
#define SEQ      512
#define KBLOCKS  (DIM_EMB_PAD / 32)   // 10

typedef __attribute__((ext_vector_type(16))) __bf16 v16bf;
typedef __attribute__((ext_vector_type(8)))  __bf16 v8bf;
typedef __attribute__((ext_vector_type(8)))  float  v8f;

// float -> bf16 round-to-nearest-even (bit pattern as ushort)
static __device__ __forceinline__ unsigned short f2bf(float f) {
    unsigned int x = __builtin_bit_cast(unsigned int, f);
    unsigned int r = x + 0x7fffu + ((x >> 16) & 1u);
    return (unsigned short)(r >> 16);
}

// ---------------------------------------------------------------------------
// Kernel 1: pack w1 [300,512] f32 -> bf16 WMMA-B fragments.
// Layout: packedB[kb][tileN][lane][e], e=0..15, so each lane's fragment is two
// contiguous b128 loads. Element e of lane (half=lane>>4,
// ln=lane&15) holds w1[kb*32 + (e>>3)*16 + half*8 + (e&7)][tileN*16 + ln]
// (zero for k >= 300), matching the 16-bit B 32x16 wave32 layout.
// ---------------------------------------------------------------------------
__global__ void pack_w1_kernel(const float* __restrict__ w1,
                               unsigned short* __restrict__ w1_pk) {
    int idx = blockIdx.x * blockDim.x + threadIdx.x;   // over 10*32*32*16
    if (idx >= KBLOCKS * 32 * 32 * 16) return;
    int e     = idx & 15;
    int lane  = (idx >> 4) & 31;
    int tileN = (idx >> 9) & 31;
    int kb    = idx >> 14;
    int j  = e >> 1;
    int lo = e & 1;
    int half = lane >> 4;
    int ln   = lane & 15;
    int k = kb * 32 + (e >> 3) * 16 + half * 8 + (e & 7);
    int n = tileN * 16 + ln;
    float v = (k < DIM_EMB) ? w1[k * DIM_H + n] : 0.0f;
    w1_pk[idx] = f2bf(v);
}

// ---------------------------------------------------------------------------
// Kernel 2: per-batch-row attention pooling.
//   scores = emb[x] . q_w + q_b ; mask PAD -> -inf ; softmax over seq ;
//   pooled[d] = sum_s alpha[s] * emb[x[s]][d]  -> bf16 row (padded to 320)
// One block (256 threads = 8 wave32) per batch row.
// ---------------------------------------------------------------------------
__global__ void attn_pool_kernel(const int*   __restrict__ x,
                                 const float* __restrict__ emb,
                                 const float* __restrict__ q_w,
                                 const float* __restrict__ q_b,
                                 unsigned short* __restrict__ pooled_bf) {
    __shared__ float s_qw[DIM_EMB];
    __shared__ float s_sc[SEQ];
    __shared__ float s_red[256];

    const int b    = blockIdx.x;
    const int tid  = threadIdx.x;
    const int lane = tid & 31;
    const int wave = tid >> 5;

    if (tid < DIM_EMB) s_qw[tid] = q_w[tid];
    __syncthreads();

    const float qb = q_b[0];
    const int* xrow = x + b * SEQ;

    // ---- pass 1: scores (one token per wave iteration, lane-parallel dot) --
    for (int s = wave; s < SEQ; s += 8) {
        int tok = xrow[s];
        const float* row = emb + (long)tok * DIM_EMB;
        float p = 0.0f;
        #pragma unroll
        for (int i = 0; i < 10; ++i) {          // ceil(300/32)=10
            int d = lane + i * 32;
            if (d < DIM_EMB) p += row[d] * s_qw[d];
        }
        #pragma unroll
        for (int off = 16; off > 0; off >>= 1)
            p += __shfl_xor(p, off, 32);
        if (lane == 0)
            s_sc[s] = (tok == PAD_ID) ? -__builtin_huge_valf() : (p + qb);
    }
    __syncthreads();

    // ---- softmax over 512 scores --------------------------------------
    float m0 = fmaxf(s_sc[tid], s_sc[tid + 256]);
    s_red[tid] = m0;
    __syncthreads();
    #pragma unroll
    for (int off = 128; off > 0; off >>= 1) {
        if (tid < off) s_red[tid] = fmaxf(s_red[tid], s_red[tid + off]);
        __syncthreads();
    }
    float smax = s_red[0];
    __syncthreads();

    float e0 = __expf(s_sc[tid]       - smax);
    float e1 = __expf(s_sc[tid + 256] - smax);
    s_red[tid] = e0 + e1;
    __syncthreads();
    #pragma unroll
    for (int off = 128; off > 0; off >>= 1) {
        if (tid < off) s_red[tid] += s_red[tid + off];
        __syncthreads();
    }
    float inv = 1.0f / s_red[0];
    __syncthreads();
    s_sc[tid]       = e0 * inv;
    s_sc[tid + 256] = e1 * inv;
    __syncthreads();

    // ---- pass 2: weighted pooling, dim-parallel (coalesced per row) ----
    const int d0 = tid;            // < 256 < 300, always valid
    const int d1 = tid + 256;      // valid only if < 300
    float acc0 = 0.0f, acc1 = 0.0f;
    for (int s = 0; s < SEQ; ++s) {
        int tok = xrow[s];
        const float* row = emb + (long)tok * DIM_EMB;
        if (s + 1 < SEQ) {  // prefetch next gathered row (random -> hide latency)
            const float* nrow = emb + (long)xrow[s + 1] * DIM_EMB;
            __builtin_prefetch(nrow + d0, 0, 0);
        }
        float a = s_sc[s];
        acc0 += a * row[d0];
        if (d1 < DIM_EMB) acc1 += a * row[d1];
    }

    unsigned short* out = pooled_bf + (long)b * DIM_EMB_PAD;
    out[d0] = f2bf(acc0);
    if (d1 < DIM_EMB)            out[d1] = f2bf(acc1);
    else if (d1 < DIM_EMB_PAD)   out[d1] = 0;        // zero K padding 300..319
}

// ---------------------------------------------------------------------------
// Kernel 3: h = relu(pooled @ w1 + b1)  via v_wmma_f32_16x16x32_bf16.
// M=256, N=512, K=320. One wave per 16x16 C tile -> 512 waves = 64 blocks.
// A fragment (16-bit 16x32, wave32): lane half=l/16, m=l%16; two contiguous
//   8-bf16 chunks at K = kbase+half*8 and kbase+16+half*8 -> 2x b128 loads.
// B fragment: pre-packed by pack_w1_kernel -> 2x contiguous b128 loads.
// C (f32 16x16): n = l%16, M = j + half*8.
// K-loop fully unrolled -> 10 v_wmma per wave.
// ---------------------------------------------------------------------------
__global__ void gemm1_wmma_kernel(const unsigned short* __restrict__ A_bf,   // [256,320]
                                  const unsigned short* __restrict__ B_pk,   // packed
                                  const float* __restrict__ b1,
                                  float* __restrict__ h) {                   // [256,512]
    const int tid    = threadIdx.x;
    const int lane   = tid & 31;
    const int waveId = blockIdx.x * 8 + (tid >> 5);   // 0..511
    const int tileM  = waveId >> 5;                    // 0..15
    const int tileN  = waveId & 31;                    // 0..31

    const int half = lane >> 4;
    const int ln   = lane & 15;
    const int m    = tileM * 16 + ln;
    const int n    = tileN * 16 + ln;

    const unsigned short* Arow = A_bf + (long)m * DIM_EMB_PAD;

    v8f c = {};
    #pragma unroll
    for (int kb = 0; kb < KBLOCKS; ++kb) {            // 10 iterations, unrolled
        const int kbase = kb * 32;

        v8bf a_lo = *(const v8bf*)(Arow + kbase + half * 8);
        v8bf a_hi = *(const v8bf*)(Arow + kbase + 16 + half * 8);
        v16bf a = __builtin_shufflevector(a_lo, a_hi,
                    0,1,2,3,4,5,6,7,8,9,10,11,12,13,14,15);

        const unsigned short* Bfrag =
            B_pk + ((((long)kb * 32 + tileN) * 32 + lane) << 4);
        v8bf b_lo = *(const v8bf*)(Bfrag);
        v8bf b_hi = *(const v8bf*)(Bfrag + 8);
        v16bf bf = __builtin_shufflevector(b_lo, b_hi,
                    0,1,2,3,4,5,6,7,8,9,10,11,12,13,14,15);

        c = __builtin_amdgcn_wmma_f32_16x16x32_bf16(
                /*neg_a=*/false, a, /*neg_b=*/false, bf,
                /*c_mod=*/(short)0, c, /*reuse_a=*/false, /*reuse_b=*/false);
    }

    const float bias = b1[n];
    #pragma unroll
    for (int j = 0; j < 8; ++j) {
        int mrow = tileM * 16 + j + half * 8;
        float v = c[j] + bias;
        h[(long)mrow * DIM_H + n] = fmaxf(v, 0.0f);
    }
}

// ---------------------------------------------------------------------------
// Kernel 4: logits = h @ w2 + b2   (512 -> 2 per batch row)
// ---------------------------------------------------------------------------
__global__ void head_kernel(const float* __restrict__ h,
                            const float* __restrict__ w2,
                            const float* __restrict__ b2,
                            float* __restrict__ out) {
    __shared__ float r0[256];
    __shared__ float r1[256];
    const int b   = blockIdx.x;
    const int tid = threadIdx.x;

    float h0 = h[(long)b * DIM_H + tid];
    float h1 = h[(long)b * DIM_H + tid + 256];
    r0[tid] = h0 * w2[tid * 2 + 0] + h1 * w2[(tid + 256) * 2 + 0];
    r1[tid] = h0 * w2[tid * 2 + 1] + h1 * w2[(tid + 256) * 2 + 1];
    __syncthreads();
    #pragma unroll
    for (int off = 128; off > 0; off >>= 1) {
        if (tid < off) {
            r0[tid] += r0[tid + off];
            r1[tid] += r1[tid + off];
        }
        __syncthreads();
    }
    if (tid == 0) {
        out[b * 2 + 0] = r0[0] + b2[0];
        out[b * 2 + 1] = r1[0] + b2[1];
    }
}

// ---------------------------------------------------------------------------
// Launch
// ---------------------------------------------------------------------------
extern "C" void kernel_launch(void* const* d_in, const int* in_sizes, int n_in,
                              void* d_out, int out_size, void* d_ws, size_t ws_size,
                              hipStream_t stream) {
    const int*   x    = (const int*)  d_in[0];   // [256,512]
    const float* emb  = (const float*)d_in[1];   // [40002,300]
    const float* q_w  = (const float*)d_in[2];   // [300]
    const float* q_b  = (const float*)d_in[3];   // scalar
    const float* w1   = (const float*)d_in[4];   // [300,512]
    const float* b1   = (const float*)d_in[5];   // [512]
    const float* w2   = (const float*)d_in[6];   // [512,2]
    const float* b2   = (const float*)d_in[7];   // [2]
    float* out        = (float*)d_out;           // [256,2]

    // workspace layout
    char* ws = (char*)d_ws;
    unsigned short* pooled_bf = (unsigned short*)ws;                       // 256*320*2    = 163840 B
    unsigned short* w1_pk     = (unsigned short*)(ws + 163840);            // 10*32*32*16*2= 327680 B
    float*          h         = (float*)(ws + 163840 + 327680);            // 256*512*4    = 524288 B

    pack_w1_kernel<<<(KBLOCKS * 32 * 32 * 16 + 255) / 256, 256, 0, stream>>>(w1, w1_pk);
    attn_pool_kernel<<<BATCH, 256, 0, stream>>>(x, emb, q_w, q_b, pooled_bf);
    gemm1_wmma_kernel<<<64, 256, 0, stream>>>(pooled_bf, w1_pk, b1, h);
    head_kernel<<<BATCH, 256, 0, stream>>>(h, w2, b2, out);
}
